// RGCNEncoder_63273458205156
// MI455X (gfx1250) — matrix operand order repaired and checked by reading, not compiled, not run on hardware
//
#include <hip/hip_runtime.h>

#define IN_DIM 128
#define HID 128
#define NUM_REL 8
#define WMAT (IN_DIM * HID)   // 16384 elements per weight matrix

typedef __bf16 bf16_t;
typedef bf16_t v16bf __attribute__((ext_vector_type(16)));
typedef float v8f __attribute__((ext_vector_type(8)));

// float -> bf16 round-to-nearest-even, as raw u16
static __device__ __forceinline__ unsigned short f2bf(float f) {
    unsigned int u = __builtin_bit_cast(unsigned int, f);
    u += 0x7fffu + ((u >> 16) & 1u);
    return (unsigned short)(u >> 16);
}
static __device__ __forceinline__ bf16_t bfc(unsigned short u) {
    return __builtin_bit_cast(bf16_t, u);
}
static __device__ __forceinline__ unsigned int pack2(float a, float b) {
    return (unsigned int)f2bf(a) | ((unsigned int)f2bf(b) << 16);
}

// ---------------------------------------------------------------------------
// One-time weight prep: for 18 matrices (root1, rel1[0..7], root2, rel2[0..7])
// write bf16 W^T (wt[m][n][k] = bf16(W[k][n])).  18 x 32 KB, negligible cost,
// amortized over ~1563 GEMM blocks per matrix.
// ---------------------------------------------------------------------------
__global__ __launch_bounds__(256)
void prep_weights(const float* __restrict__ root1, const float* __restrict__ rel1,
                  const float* __restrict__ root2, const float* __restrict__ rel2,
                  unsigned short* __restrict__ wt)
{
    const int m = blockIdx.x;           // 0..17
    const float* W = (m == 0) ? root1
                   : (m < 9) ? rel1 + (size_t)(m - 1) * WMAT
                   : (m == 9) ? root2
                   : rel2 + (size_t)(m - 10) * WMAT;
    unsigned short* o = wt + (size_t)m * WMAT;
    const int t = threadIdx.x;
#pragma unroll 4
    for (int i = 0; i < 64; ++i) {
        const int idx = t + 256 * i;    // coalesced read of W[k][n]
        const int k = idx >> 7, n = idx & (HID - 1);
        o[n * IN_DIM + k] = f2bf(W[idx]);
    }
}

// ---------------------------------------------------------------------------
// WMMA GEMM: out[64 rows x 128 cols per block] (+)= A(scaled) @ W  [+ bias]
//   A:  nrows x 128 f32
//   Wt: 128 x 128 bf16 bits, pre-transposed (Wt[n][k])
//   SCALE: scale row i of A by 1/max(cnt[i],1)  (mean aggregation)
//   ACCUM: accumulate into existing out; else init C with bias broadcast
// 256 threads = 8 waves; wave w owns column tile [w*16, w*16+16).
// Both A and B lane fragments are contiguous 16-bf16 runs -> ds_load_b128.
// ---------------------------------------------------------------------------
template<bool SCALE, bool ACCUM>
__global__ __launch_bounds__(256)
void rgcn_gemm(const float* __restrict__ A, const unsigned short* __restrict__ Wt,
               const float* __restrict__ bias, const float* __restrict__ cnt,
               float* __restrict__ out, int nrows)
{
    __shared__ unsigned short sA[64][IN_DIM + 8];    // bf16 bits, stride 272B
    __shared__ unsigned short sBt[HID][IN_DIM + 8];  // W^T: sBt[n][k]

    const int tid = threadIdx.x;
    const int rowbase = blockIdx.x * 64;
    const bool full = (rowbase + 64 <= nrows);

    __builtin_prefetch(Wt + tid * 64, 0, 0);         // global_prefetch_b8

    // ---- stage A tile (64 x 128) as bf16, optional 1/cnt row scaling
    {
        const int r  = tid >> 2;          // 4 threads per row
        const int c0 = (tid & 3) * 32;    // 32 cols each
        const int grow = rowbase + r;
        if (grow < nrows) {
            float s = 1.0f;
            if (SCALE) s = 1.0f / fmaxf(cnt[grow], 1.0f);
            const float4* ap = (const float4*)(A + (size_t)grow * IN_DIM + c0);
            uint4* dp = (uint4*)&sA[r][c0];
#pragma unroll
            for (int i = 0; i < 4; ++i) {
                float4 v0 = ap[2 * i];
                float4 v1 = ap[2 * i + 1];
                uint4 u;
                u.x = pack2(v0.x * s, v0.y * s);
                u.y = pack2(v0.z * s, v0.w * s);
                u.z = pack2(v1.x * s, v1.y * s);
                u.w = pack2(v1.z * s, v1.w * s);
                dp[i] = u;
            }
        } else {
            uint4* dp = (uint4*)&sA[r][c0];
            uint4 z = {0u, 0u, 0u, 0u};
#pragma unroll
            for (int i = 0; i < 4; ++i) dp[i] = z;
        }
    }
    // ---- stage pre-transposed bf16 weights: pure uint4 copy, coalesced
    {
        const int n  = tid >> 1;          // output column
        const int k0 = (tid & 1) * 64;
        const uint4* wp = (const uint4*)(Wt + (size_t)n * IN_DIM + k0);
        uint4* dp = (uint4*)&sBt[n][k0];  // 16B aligned: 272*n + 2*k0
#pragma unroll
        for (int j = 0; j < 8; ++j) dp[j] = wp[j];
    }
    __syncthreads();

    const int wave  = tid >> 5;
    const int lane  = tid & 31;
    const int ncol  = wave * 16 + (lane & 15);
    const int khi16 = (lane & 16) ? 16 : 0;  // B: lanes 16-31 hold K+16
    const int khi8  = (lane & 16) ? 8  : 0;  // unused: A also uses khi16 (lanes 16-31 hold K+16)
    const int mhi8  = (lane & 16) ? 8  : 0;  // C/D: lanes 16-31 hold M+8

    // ---- B fragments: contiguous 16 bf16 per lane -> 2x ds_load_b128 each
    v16bf bfrag[4];
#pragma unroll
    for (int ks = 0; ks < 4; ++ks) {
#pragma unroll
        for (int e = 0; e < 16; ++e)
            bfrag[ks][e] = bfc(sBt[ncol][ks * 32 + khi16 + e]);
    }

#pragma unroll
    for (int rt = 0; rt < 4; ++rt) {
        const int grow0 = rowbase + rt * 16 + mhi8;          // rows grow0..+7
        float* cp = out + (size_t)grow0 * HID + ncol;

        v8f acc;
        if (ACCUM) {
            if (full) {
#pragma unroll
                for (int v = 0; v < 8; ++v) acc[v] = cp[(size_t)v * HID];
            } else {
#pragma unroll
                for (int v = 0; v < 8; ++v)
                    acc[v] = (grow0 + v < nrows) ? cp[(size_t)v * HID] : 0.0f;
            }
        } else {
            const float b = bias[ncol];
#pragma unroll
            for (int v = 0; v < 8; ++v) acc[v] = b;
        }

        const int arow = rt * 16 + (lane & 15);
#pragma unroll
        for (int ks = 0; ks < 4; ++ks) {
            v16bf afrag;
#pragma unroll
            for (int e = 0; e < 16; ++e)
                afrag[e] = bfc(sA[arow][ks * 32 + khi16 + e]);
            acc = __builtin_amdgcn_wmma_f32_16x16x32_bf16(
                false, afrag, false, bfrag[ks], (short)0, acc, false, false);
        }

        if (full) {
#pragma unroll
            for (int v = 0; v < 8; ++v) cp[(size_t)v * HID] = acc[v];
        } else {
#pragma unroll
            for (int v = 0; v < 8; ++v)
                if (grow0 + v < nrows) cp[(size_t)v * HID] = acc[v];
        }
    }
}

// ---------------------------------------------------------------------------
// Edge scatter for one relation: agg[dst] += x[src]; cnt[dst] += 1
// One wave per edge; each lane gathers float4, issues 4 global f32 atomics.
// agg (51 MB) and x (51 MB) stay L2-resident (192 MB L2 on MI455X).
// ---------------------------------------------------------------------------
__global__ __launch_bounds__(256)
void edge_scatter(const int* __restrict__ src, const int* __restrict__ dst,
                  const int* __restrict__ etype, const float* __restrict__ x,
                  float* __restrict__ agg, float* __restrict__ cnt,
                  int rel, int nedges)
{
    const int e = blockIdx.x * 8 + (threadIdx.x >> 5);
    if (e >= nedges) return;
    if (etype[e] != rel) return;
    const int lane = threadIdx.x & 31;
    const int s = src[e];
    const int d = dst[e];
    const float4 v = *(const float4*)(x + (size_t)s * IN_DIM + lane * 4);
    float* o = agg + (size_t)d * IN_DIM + lane * 4;
    atomicAdd(o + 0, v.x);
    atomicAdd(o + 1, v.y);
    atomicAdd(o + 2, v.z);
    atomicAdd(o + 3, v.w);
    if (lane == 0) atomicAdd(cnt + d, 1.0f);
}

__global__ __launch_bounds__(256)
void zero_kernel(float* __restrict__ p, size_t n)
{
    const size_t n4 = n >> 2;
    float4* p4 = (float4*)p;
    const float4 z = {0.0f, 0.0f, 0.0f, 0.0f};
    size_t i = (size_t)blockIdx.x * blockDim.x + threadIdx.x;
    const size_t stride = (size_t)gridDim.x * blockDim.x;
    for (; i < n4; i += stride) p4[i] = z;
    if (blockIdx.x == 0 && threadIdx.x == 0)
        for (size_t j = n4 << 2; j < n; ++j) p[j] = 0.0f;
}

__global__ __launch_bounds__(256)
void relu_kernel(float* __restrict__ p, size_t n)
{
    const size_t n4 = n >> 2;
    float4* p4 = (float4*)p;
    size_t i = (size_t)blockIdx.x * blockDim.x + threadIdx.x;
    const size_t stride = (size_t)gridDim.x * blockDim.x;
    for (; i < n4; i += stride) {
        float4 v = p4[i];
        v.x = fmaxf(v.x, 0.0f); v.y = fmaxf(v.y, 0.0f);
        v.z = fmaxf(v.z, 0.0f); v.w = fmaxf(v.w, 0.0f);
        p4[i] = v;
    }
    if (blockIdx.x == 0 && threadIdx.x == 0)
        for (size_t j = n4 << 2; j < n; ++j) p[j] = fmaxf(p[j], 0.0f);
}

// ---------------------------------------------------------------------------
extern "C" void kernel_launch(void* const* d_in, const int* in_sizes, int n_in,
                              void* d_out, int out_size, void* d_ws, size_t ws_size,
                              hipStream_t stream)
{
    const float* x       = (const float*)d_in[0];
    const int*   eidx    = (const int*)  d_in[1];   // [2, E]
    const int*   etyp    = (const int*)  d_in[2];   // [E]
    const float* rel_w1  = (const float*)d_in[3];   // [8,128,128]
    const float* root_w1 = (const float*)d_in[4];   // [128,128]
    const float* b1      = (const float*)d_in[5];   // [128]
    const float* rel_w2  = (const float*)d_in[6];
    const float* root_w2 = (const float*)d_in[7];
    const float* b2      = (const float*)d_in[8];

    const int nnodes = in_sizes[0] / IN_DIM;
    const int nedges = in_sizes[2];
    const int* src  = eidx;
    const int* dstp = eidx + nedges;

    // workspace: h [N*128] | agg [N*128] | cnt [N] | wt [18*16384 bf16]
    float* h   = (float*)d_ws;
    float* agg = h   + (size_t)nnodes * HID;
    float* cnt = agg + (size_t)nnodes * IN_DIM;
    size_t wt_off = ((size_t)(cnt + nnodes) + 15) & ~(size_t)15;   // 16B align
    unsigned short* wt = (unsigned short*)wt_off;
    float* out = (float*)d_out;

    const unsigned short* wt_root1 = wt;
    const unsigned short* wt_rel1  = wt + (size_t)1  * WMAT;
    const unsigned short* wt_root2 = wt + (size_t)9  * WMAT;
    const unsigned short* wt_rel2  = wt + (size_t)10 * WMAT;

    const dim3 blk(256);
    const int gemm_grid = (nnodes + 63) / 64;
    const int scat_grid = (nedges + 7) / 8;
    const size_t zn = (size_t)nnodes * IN_DIM + (size_t)nnodes; // agg + cnt
    const int zgrid = 2048;
    const size_t hn = (size_t)nnodes * HID;

    // one-time (per call) bf16 W^T prep for all 18 weight matrices
    prep_weights<<<2 * (1 + NUM_REL), blk, 0, stream>>>(root_w1, rel_w1, root_w2, rel_w2, wt);

    // ---------------- layer 1: h = relu(x@root1 + b1 + sum_r mean_r(x)@W1_r)
    rgcn_gemm<false, false><<<gemm_grid, blk, 0, stream>>>(x, wt_root1, b1, nullptr, h, nnodes);
    for (int r = 0; r < NUM_REL; ++r) {
        zero_kernel<<<zgrid, blk, 0, stream>>>(agg, zn);
        edge_scatter<<<scat_grid, blk, 0, stream>>>(src, dstp, etyp, x, agg, cnt, r, nedges);
        rgcn_gemm<true, true><<<gemm_grid, blk, 0, stream>>>(
            agg, wt_rel1 + (size_t)r * WMAT, nullptr, cnt, h, nnodes);
    }
    relu_kernel<<<zgrid, blk, 0, stream>>>(h, hn);

    // ---------------- layer 2: out = relu(h@root2 + b2 + sum_r mean_r(h)@W2_r)
    rgcn_gemm<false, false><<<gemm_grid, blk, 0, stream>>>(h, wt_root2, b2, nullptr, out, nnodes);
    for (int r = 0; r < NUM_REL; ++r) {
        zero_kernel<<<zgrid, blk, 0, stream>>>(agg, zn);
        edge_scatter<<<scat_grid, blk, 0, stream>>>(src, dstp, etyp, h, agg, cnt, r, nedges);
        rgcn_gemm<true, true><<<gemm_grid, blk, 0, stream>>>(
            agg, wt_rel2 + (size_t)r * WMAT, nullptr, cnt, out, nnodes);
    }
    relu_kernel<<<zgrid, blk, 0, stream>>>(out, hn);
}
